// PatchedLmHeadLinearAllreduce_10333691314386
// MI455X (gfx1250) — 8-bit matrix operand order repaired and checked by reading, compiled, not run on hardware
//
#include <hip/hip_runtime.h>
#include <hip/hip_bf16.h>
#include <math.h>


typedef __attribute__((ext_vector_type(16))) int   v16i;
typedef __attribute__((ext_vector_type(8)))  float v8f;

#define M_DIM 4096
#define N_DIM 32000
#define K_DIM 4096
#define KT    (K_DIM / 128)
#define INV_SCALE_IN 2.0f
#define INV_SCALE_W  4.0f
#define DEQUANT      0.125f

#define WMMA_FP8(a, b, c) \
  __builtin_amdgcn_wmma_f32_16x16x128_fp8_fp8((a), (b), (short)0, (c), false, false)

__device__ inline unsigned to_e4m3_sw(float x) {
  union { float f; unsigned u; } a; a.f = x;
  unsigned sign = (a.u >> 24) & 0x80u;
  unsigned absu = a.u & 0x7fffffffu;
  if (absu >= 0x7f800000u) return sign | 0x7fu;
  int      e    = (int)(absu >> 23) - 127;
  unsigned mant = absu & 0x7fffffu;
  if (e < -6) {
    union { float f; unsigned u; } b; b.u = absu;
    int q = (int)rintf(b.f * 512.0f);
    if (q >= 8) return sign | 0x08u;
    return sign | (unsigned)q;
  }
  unsigned m3   = mant >> 20;
  unsigned rem  = mant & 0xfffffu;
  if (rem > 0x80000u || (rem == 0x80000u && (m3 & 1u))) ++m3;
  if (m3 == 8u) { m3 = 0u; ++e; }
  if (e > 8) return sign | 0x7eu;
  return sign | ((unsigned)(e + 7) << 3) | m3;
}

__device__ inline unsigned pack4_e4m3(float4 f, float s) {
#if __has_builtin(__builtin_amdgcn_cvt_pk_fp8_f32)
  int w = __builtin_amdgcn_cvt_pk_fp8_f32(f.x * s, f.y * s, 0, false);
  w     = __builtin_amdgcn_cvt_pk_fp8_f32(f.z * s, f.w * s, w, true);
  return (unsigned)w;
#else
  return  to_e4m3_sw(f.x * s)        | (to_e4m3_sw(f.y * s) << 8) |
         (to_e4m3_sw(f.z * s) << 16) | (to_e4m3_sw(f.w * s) << 24);
#endif
}

__global__ __launch_bounds__(256) void quant_input_e4m3(
    const float* __restrict__ src, unsigned* __restrict__ dst) {
  unsigned d      = blockIdx.x * 256u + threadIdx.x;
  unsigned blk    = d >> 9;
  unsigned within = d & 511u;
  unsigned lane   = within >> 4;
  unsigned v      = within & 15u;
  unsigned mtile  = blk / KT;
  unsigned kt     = blk % KT;
  unsigned m      = mtile * 16u + (lane & 15u);
  unsigned hi     = lane >> 4;
  unsigned half   = v >> 3;
  unsigned v8     = v & 7u;
  unsigned k      = kt * 128u + (v >> 2) * 32u + hi * 16u + (v & 3u) * 4u;
  const float4 f  = *(const float4*)(src + (size_t)m * K_DIM + k);
  dst[d] = pack4_e4m3(f, INV_SCALE_IN);
}

__global__ __launch_bounds__(256) void quant_weight_e4m3(
    const float* __restrict__ src, unsigned* __restrict__ dst) {
  unsigned d      = blockIdx.x * 256u + threadIdx.x;
  unsigned blk    = d >> 9;
  unsigned within = d & 511u;
  unsigned lane   = within >> 4;
  unsigned v      = within & 15u;
  unsigned ntile  = blk / KT;
  unsigned kt     = blk % KT;
  unsigned n      = ntile * 16u + (lane & 15u);
  unsigned khalf  = lane >> 4;
  unsigned k      = kt * 128u + (v >> 2) * 32u + khalf * 16u + (v & 3u) * 4u;
  const float4 f  = *(const float4*)(src + (size_t)n * K_DIM + k);
  dst[d] = pack4_e4m3(f, INV_SCALE_W);
}

__global__ __launch_bounds__(256) void lmhead_fp8_gemm(
    const unsigned char* __restrict__ qa,
    const unsigned char* __restrict__ qw,
    const float* __restrict__ bias,
    float* __restrict__ out) {
  const int lane   = threadIdx.x & 31;
  const int wave   = threadIdx.x >> 5;
  const int mtile0 = blockIdx.x * 8 + (wave & 3) * 2;
  const int ntile0 = blockIdx.y * 8 + (wave >> 2) * 4;

  const v16i* A0 = (const v16i*)qa + (size_t)(mtile0 + 0) * KT * 32 + lane;
  const v16i* A1 = (const v16i*)qa + (size_t)(mtile0 + 1) * KT * 32 + lane;
  const v16i* B0 = (const v16i*)qw + (size_t)(ntile0 + 0) * KT * 32 + lane;
  const v16i* B1 = (const v16i*)qw + (size_t)(ntile0 + 1) * KT * 32 + lane;
  const v16i* B2 = (const v16i*)qw + (size_t)(ntile0 + 2) * KT * 32 + lane;
  const v16i* B3 = (const v16i*)qw + (size_t)(ntile0 + 3) * KT * 32 + lane;

  v8f acc00 = {}, acc01 = {}, acc02 = {}, acc03 = {};
  v8f acc10 = {}, acc11 = {}, acc12 = {}, acc13 = {};

  v16i a0 = A0[0], a1 = A1[0];
  v16i b0 = B0[0], b1 = B1[0], b2 = B2[0], b3 = B3[0];

#pragma unroll 2
  for (int kt = 0; kt < KT - 1; ++kt) {
    const int nx = (kt + 1) * 32;
    v16i a0n = A0[nx];
    v16i a1n = A1[nx];
    v16i b0n = B0[nx];
    v16i b1n = B1[nx];
    v16i b2n = B2[nx];
    v16i b3n = B3[nx];
    acc00 = WMMA_FP8(a0, b0, acc00);
    acc10 = WMMA_FP8(a1, b0, acc10);
    acc01 = WMMA_FP8(a0, b1, acc01);
    acc11 = WMMA_FP8(a1, b1, acc11);
    acc02 = WMMA_FP8(a0, b2, acc02);
    acc12 = WMMA_FP8(a1, b2, acc12);
    acc03 = WMMA_FP8(a0, b3, acc03);
    acc13 = WMMA_FP8(a1, b3, acc13);
    a0 = a0n; a1 = a1n; b0 = b0n; b1 = b1n; b2 = b2n; b3 = b3n;
  }
  acc00 = WMMA_FP8(a0, b0, acc00);
  acc10 = WMMA_FP8(a1, b0, acc10);
  acc01 = WMMA_FP8(a0, b1, acc01);
  acc11 = WMMA_FP8(a1, b1, acc11);
  acc02 = WMMA_FP8(a0, b2, acc02);
  acc12 = WMMA_FP8(a1, b2, acc12);
  acc03 = WMMA_FP8(a0, b3, acc03);
  acc13 = WMMA_FP8(a1, b3, acc13);

  const int ncol   = lane & 15;
  const int rowoff = (lane >> 4) << 3;
  const int n0 = (ntile0 + 0) * 16 + ncol;
  const int n1 = (ntile0 + 1) * 16 + ncol;
  const int n2 = (ntile0 + 2) * 16 + ncol;
  const int n3 = (ntile0 + 3) * 16 + ncol;
  const float bv0 = bias[n0];
  const float bv1 = bias[n1];
  const float bv2 = bias[n2];
  const float bv3 = bias[n3];

#pragma unroll
  for (int j = 0; j < 8; ++j) {
    float* r0 = out + (size_t)((mtile0 + 0) * 16 + rowoff + j) * N_DIM;
    float* r1 = out + (size_t)((mtile0 + 1) * 16 + rowoff + j) * N_DIM;
    r0[n0] = acc00[j] * DEQUANT + bv0;
    r0[n1] = acc01[j] * DEQUANT + bv1;
    r0[n2] = acc02[j] * DEQUANT + bv2;
    r0[n3] = acc03[j] * DEQUANT + bv3;
    r1[n0] = acc10[j] * DEQUANT + bv0;
    r1[n1] = acc11[j] * DEQUANT + bv1;
    r1[n2] = acc12[j] * DEQUANT + bv2;
    r1[n3] = acc13[j] * DEQUANT + bv3;
  }
}

extern "C" void kernel_launch(void* const* d_in, const int* in_sizes, int n_in,
                              void* d_out, int out_size, void* d_ws, size_t ws_size,
                              hipStream_t stream) {
  const float* input  = (const float*)d_in[0];
  const float* weight = (const float*)d_in[1];
  const float* bias   = (const float*)d_in[2];
  float*       out    = (float*)d_out;

  unsigned char* qa = (unsigned char*)d_ws;
  unsigned char* qw = qa + (size_t)M_DIM * K_DIM;

  quant_input_e4m3<<<(M_DIM * K_DIM / 4) / 256, 256, 0, stream>>>(
      input, (unsigned*)qa);
  quant_weight_e4m3<<<((size_t)N_DIM * K_DIM / 4) / 256, 256, 0, stream>>>(
      weight, (unsigned*)qw);

  dim3 grid(M_DIM / 128, N_DIM / 128);
  lmhead_fp8_gemm<<<grid, 256, 0, stream>>>(qa, qw, bias, out);
}
